// Network_9809705304943
// MI455X (gfx1250) — matrix operand order repaired and checked by reading, not compiled, not run on hardware
//
#include <hip/hip_runtime.h>

// ---------------------------------------------------------------------------
// Problem constants (match the JAX reference)
// ---------------------------------------------------------------------------
#define NN    4096   // N
#define NA    3072   // NA0
#define BB    32     // batch
#define TT    50     // timesteps
#define KT_AB 128    // 4096/32 K-tiles of Wab
#define KT_ST 96     // 3072/32 K-tiles of Wstp

#define EXP_DT_TAU       0.99501247919268232438f  // exp(-0.1/20)
#define DT_TAU           0.005f
#define EXP_DT_TAU_SYN   0.95122942450071400910f  // exp(-0.1/2)
#define DT_TAU_SYN       0.05f
#define EXP_DT_TAU_NMDA  0.99875078092458093546f  // exp(-0.1/80)
#define DT_TAU_NMDA      0.00125f

typedef __attribute__((ext_vector_type(16))) __bf16        v16bf;
typedef __attribute__((ext_vector_type(8)))  float         v8f;
typedef __attribute__((ext_vector_type(4)))  unsigned int  v4u;
typedef __attribute__((ext_vector_type(8)))  int           v8i;
typedef __attribute__((ext_vector_type(4)))  int           v4i;

struct __align__(16) U4 { unsigned a, b, c, d; };
struct __align__(32) U8 { U4 lo, hi; };

#if __has_builtin(__builtin_amdgcn_tensor_load_to_lds)
#define HAVE_TDM 1
#else
#define HAVE_TDM 0
#endif

// float -> bf16, round-to-nearest-even
__device__ __forceinline__ unsigned short f2bf(float f) {
    union { float f; unsigned u; } v; v.f = f;
    unsigned r = v.u + 0x7FFFu + ((v.u >> 16) & 1u);
    return (unsigned short)(r >> 16);
}

// ---------------------------------------------------------------------------
// One-time: pack fp32 weight matrix [K x J] (row-major) into bf16 WMMA
// B-operand tiles (B tile = 32(K) x 16(J), lane-contiguous 32B):
//   lane = (j%16) + 16*(((k%32)>>3)&1),  half = (k%8) + 8*((k%32)>>4)
// ---------------------------------------------------------------------------
__global__ void pack_weights(const float* __restrict__ W,
                             unsigned short* __restrict__ Wp,
                             int K, int J, int KT) {
    long long idx = (long long)blockIdx.x * blockDim.x + threadIdx.x;
    if (idx >= (long long)K * J) return;
    int k = (int)(idx / J);
    int j = (int)(idx % J);
    int kt = k >> 5, jt = j >> 4, kk = k & 31;
    int lane = (j & 15) + 16 * ((kk >> 3) & 1);
    int half = (kk & 7) + 8 * (kk >> 4);
    Wp[(((size_t)jt * KT + kt) * 32 + lane) * 16 + half] = f2bf(W[idx]);
}

// ---------------------------------------------------------------------------
// Initial state
// ---------------------------------------------------------------------------
__global__ void init_state(const float* __restrict__ ff,
                           const float* __restrict__ rec_init,
                           float* __restrict__ rates,
                           float* __restrict__ rec0,
                           float* __restrict__ rec1,
                           float* __restrict__ u,
                           float* __restrict__ x) {
    int idx = blockIdx.x * blockDim.x + threadIdx.x;
    if (idx >= BB * NN) return;
    int b = idx / NN, n = idx % NN;
    float r0 = rec_init[idx];
    float r1 = rec_init[(size_t)BB * NN + idx];
    float f0 = ff[(size_t)b * TT * NN + n];          // t = 0
    rates[idx] = fmaxf(f0 + r0 - 1.0f, 0.0f);
    rec0[idx] = r0;
    rec1[idx] = r1;
    if (n < NA) {
        size_t si = (size_t)b * NA + n;
        u[si] = 0.03f;
        x[si] = 1.0f;
    }
}

// ---------------------------------------------------------------------------
// Per-step: STP update + pack activations into bf16 WMMA A-operand tiles.
// A tile = 16(M) x 32(K), stored as two 512B half-tiles so LDS reads are
// 16B/lane stride (bank-conflict-free ds_load_b128):
//   kk=k%32; hi=kk>>3; lane=(m%16)+16*(hi&1); chunk=hi>>1; h=kk&7
//   u16 offset within 512-elem tile = chunk*256 + lane*8 + h
// ---------------------------------------------------------------------------
__global__ void stp_and_pack(const float* __restrict__ rates,
                             float* __restrict__ u,
                             float* __restrict__ x,
                             unsigned short* __restrict__ Ar,   // rates, KT=128
                             unsigned short* __restrict__ Aa) { // aux,   KT=96
    int idx = blockIdx.x * blockDim.x + threadIdx.x;
    if (idx >= BB * NN) return;
    int m = idx / NN, k = idx % NN;
    float re = rates[idx];

    int mt = m >> 4, kt = k >> 5, kk = k & 31, hi = kk >> 3;
    int lane = (m & 15) + 16 * (hi & 1);
    unsigned off = (unsigned)((hi >> 1) * 256 + lane * 8 + (kk & 7));

    Ar[(size_t)(mt * KT_AB + kt) * 512 + off] = f2bf(re);

    if (k < NA) {
        size_t si = (size_t)m * NA + k;
        float uu = u[si], xx = x[si];
        float un = uu + 0.1f * (0.03f - uu) + 0.003f * (1.0f - uu) * re;
        float xn = xx + 0.4f * (1.0f - xx) - 0.1f * un * xx * re;
        u[si] = un;
        x[si] = xn;
        float aux = un * xn * re;
        Aa[(size_t)(mt * KT_ST + kt) * 512 + off] = f2bf(aux);
    }
}

// ---------------------------------------------------------------------------
// TDM: 1-D flat copy global -> LDS (bytes multiple of 8, <= 512KB).
// D# per CDNA5 ISA: group0 {count=1, lds_addr,
// global_addr, type=2}; group1 {data_size=8B, tensor_dim0=n, dim1=1,
// tile_dim0=n, stride0=n}.
// ---------------------------------------------------------------------------
#if HAVE_TDM
__device__ __forceinline__ void tdm_load_1d(const void* gsrc, unsigned lds_byte,
                                            unsigned bytes) {
    unsigned long long ga = (unsigned long long)gsrc;
    unsigned n = bytes >> 3;                                     // 8B elements
    v4u g0;
    g0[0] = 1u;                                                  // count=1
    g0[1] = lds_byte;                                            // lds_addr
    g0[2] = (unsigned)ga;                                        // ga[31:0]
    g0[3] = (unsigned)((ga >> 32) & 0x01FFFFFFu) | (2u << 30);   // ga[56:32]|type=2
    v8i g1;
    g1[0] = (int)(3u << 16);                                     // data_size=8B
    g1[1] = (int)((n & 0xFFFFu) << 16);                          // tensor_dim0 lo
    g1[2] = (int)((n >> 16) | (1u << 16));                       // dim0 hi | dim1=1
    g1[3] = (int)((n & 0xFFFFu) << 16);                          // tile_dim0 = n
    g1[4] = 0;                                                   // tile_dim1/2 = 0
    g1[5] = (int)n;                                              // dim0_stride lo
    g1[6] = 0;
    g1[7] = 0;
    v4i g2 = {0, 0, 0, 0};
    v4i g3 = {0, 0, 0, 0};
#if __has_include(<hip/amd_detail/amd_gfx1250_TDM.h>)
    v8i g4 = {0, 0, 0, 0, 0, 0, 0, 0};
    __builtin_amdgcn_tensor_load_to_lds(g0, g1, g2, g3, g4, 0);
#else
    __builtin_amdgcn_tensor_load_to_lds(g0, g1, g2, g3, 0);
#endif
}
#endif

// Read one A tile (16x32 bf16) from LDS: two conflict-free ds_load_b128.
__device__ __forceinline__ v16bf ldsA(const char* smem, unsigned tile, int lane) {
    const U4* p = (const U4*)(smem + (size_t)tile * 1024);
    U8 t;
    t.lo = p[lane];
    t.hi = p[32 + lane];
    return __builtin_bit_cast(v16bf, t);
}

// ---------------------------------------------------------------------------
// Per-step GEMM + fused epilogue.
// Block = 4 waves = 4 j-tiles; whole A operand (256KB) staged in LDS via TDM
// and shared by the block.  Exploits Wab[:NA,:NA] == 0:
//   j <  NA:  hidden = S2 + J*S3,  hidden_n = J*S3
//   j >= NA:  hidden = S1 + S2,    hidden_n = S1
// ---------------------------------------------------------------------------
__global__ __launch_bounds__(128) void gemm_step(
    const unsigned short* __restrict__ Wab_p,
    const unsigned short* __restrict__ Wstp_p,
    const unsigned short* __restrict__ Ar,
    const unsigned short* __restrict__ Aa,
    const float* __restrict__ ff,
    const float* __restrict__ Jstp,
    float* __restrict__ rates,
    float* __restrict__ rec0,
    float* __restrict__ rec1,
    float* __restrict__ out,
    int t) {
    extern __shared__ char smem[];                 // 256KB dynamic LDS
    const int jt   = blockIdx.x * 4 + (threadIdx.x >> 5);
    const int lane = threadIdx.x & 31;
    const int j0   = jt * 16;
    const bool regA = (j0 < NA);

    // ---- Stage the full A operand for this region into LDS ----
#if HAVE_TDM
    if (threadIdx.x < 32) {                        // one TDM issue per block
        unsigned lds0 = (unsigned)(unsigned long long)(void*)smem;
        if (regA) {
            // aux A: 192 tiles (2 x 96) = 196608B at LDS tile 0
            tdm_load_1d(Aa, lds0, 192 * 1024);
            // rates A, mt=0, kt 96..127 -> LDS tiles 192..223
            tdm_load_1d((const char*)Ar + 96 * 1024, lds0 + 192 * 1024, 32 * 1024);
            // rates A, mt=1, kt 96..127 -> LDS tiles 224..255
            tdm_load_1d((const char*)Ar + 224 * 1024, lds0 + 224 * 1024, 32 * 1024);
        } else {
            // full rates A: 256 tiles = 262144B
            tdm_load_1d(Ar, lds0, 256 * 1024);
        }
#if __has_builtin(__builtin_amdgcn_s_wait_tensorcnt)
        __builtin_amdgcn_s_wait_tensorcnt(0);
#else
        asm volatile("s_wait_tensorcnt 0" ::: "memory");
#endif
    }
#else
    {   // fallback: cooperative vector copy
        U4* d = (U4*)smem;
        if (regA) {
            const U4* s0 = (const U4*)Aa;
            for (int i = threadIdx.x; i < 192 * 64; i += 128) d[i] = s0[i];
            const U4* s1 = (const U4*)(Ar + (size_t)96 * 512);
            for (int i = threadIdx.x; i < 32 * 64; i += 128) d[192 * 64 + i] = s1[i];
            const U4* s2 = (const U4*)(Ar + (size_t)224 * 512);
            for (int i = threadIdx.x; i < 32 * 64; i += 128) d[224 * 64 + i] = s2[i];
        } else {
            const U4* s0 = (const U4*)Ar;
            for (int i = threadIdx.x; i < 256 * 64; i += 128) d[i] = s0[i];
        }
    }
#endif
    __syncthreads();

    v8f accP0 = {}, accP1 = {}, accQ0 = {}, accQ1 = {};
    const v16bf* Bab = (const v16bf*)Wab_p + (size_t)jt * KT_AB * 32;

    if (regA) {
        // ---- S3: aux @ Wstp, K in [0,3072) ----
        const v16bf* Bst = (const v16bf*)Wstp_p + (size_t)jt * KT_ST * 32;
        #pragma unroll 4
        for (int kt = 0; kt < KT_ST; ++kt) {
            __builtin_prefetch(&Bst[(kt + 8) * 32 + lane], 0, 1);
            v16bf b  = Bst[kt * 32 + lane];
            v16bf a0 = ldsA(smem, kt, lane);            // mt=0 aux
            v16bf a1 = ldsA(smem, KT_ST + kt, lane);    // mt=1 aux
            accP0 = __builtin_amdgcn_wmma_f32_16x16x32_bf16(
                false, a0, false, b, (short)0, accP0, false, false);
            accP1 = __builtin_amdgcn_wmma_f32_16x16x32_bf16(
                false, a1, false, b, (short)0, accP1, false, false);
        }
        // ---- S2: rates @ Wab, K in [3072,4096) ----
        #pragma unroll 4
        for (int kt = KT_ST; kt < KT_AB; ++kt) {
            v16bf b  = Bab[kt * 32 + lane];
            v16bf a0 = ldsA(smem, 192 + (kt - KT_ST), lane);
            v16bf a1 = ldsA(smem, 224 + (kt - KT_ST), lane);
            accQ0 = __builtin_amdgcn_wmma_f32_16x16x32_bf16(
                false, a0, false, b, (short)0, accQ0, false, false);
            accQ1 = __builtin_amdgcn_wmma_f32_16x16x32_bf16(
                false, a1, false, b, (short)0, accQ1, false, false);
        }
    } else {
        // ---- S1: rates @ Wab, K in [0,3072) ----
        #pragma unroll 4
        for (int kt = 0; kt < KT_ST; ++kt) {
            __builtin_prefetch(&Bab[(kt + 8) * 32 + lane], 0, 1);
            v16bf b  = Bab[kt * 32 + lane];
            v16bf a0 = ldsA(smem, kt, lane);
            v16bf a1 = ldsA(smem, KT_AB + kt, lane);
            accP0 = __builtin_amdgcn_wmma_f32_16x16x32_bf16(
                false, a0, false, b, (short)0, accP0, false, false);
            accP1 = __builtin_amdgcn_wmma_f32_16x16x32_bf16(
                false, a1, false, b, (short)0, accP1, false, false);
        }
        // ---- S2: rates @ Wab, K in [3072,4096) ----
        #pragma unroll 4
        for (int kt = KT_ST; kt < KT_AB; ++kt) {
            v16bf b  = Bab[kt * 32 + lane];
            v16bf a0 = ldsA(smem, kt, lane);
            v16bf a1 = ldsA(smem, KT_AB + kt, lane);
            accQ0 = __builtin_amdgcn_wmma_f32_16x16x32_bf16(
                false, a0, false, b, (short)0, accQ0, false, false);
            accQ1 = __builtin_amdgcn_wmma_f32_16x16x32_bf16(
                false, a1, false, b, (short)0, accQ1, false, false);
        }
    }

    // ---- Fused epilogue straight from the C/D register layout ----
    const float J = Jstp[0];
    const int n  = j0 + (lane & 15);
    const int mb = (lane >> 4) * 8;

    #pragma unroll
    for (int mt = 0; mt < 2; ++mt) {
        v8f hP = mt ? accP1 : accP0;
        v8f hQ = mt ? accQ1 : accQ0;
        #pragma unroll
        for (int r = 0; r < 8; ++r) {
            int b = mt * 16 + mb + r;
            float hidden, hidden_n;
            if (regA) {
                float s3 = J * hP[r];
                hidden   = hQ[r] + s3;
                hidden_n = s3;
            } else {
                hidden   = hP[r] + hQ[r];
                hidden_n = hP[r];
            }
            size_t idx = (size_t)b * NN + n;
            float r0 = rec0[idx] * EXP_DT_TAU_SYN  + hidden * DT_TAU_SYN;
            float r1 = rec1[idx] * EXP_DT_TAU_NMDA + 0.5f * hidden_n * DT_TAU_NMDA;
            float net = ff[((size_t)b * TT + t) * NN + n] + r0 + r1;
            float rt  = rates[idx] * EXP_DT_TAU + fmaxf(net - 1.0f, 0.0f) * DT_TAU;
            rec0[idx] = r0;
            rec1[idx] = r1;
            rates[idx] = rt;
            out[((size_t)b * TT + t) * NN + n] = rt;
        }
    }
}

// ---------------------------------------------------------------------------
// Host-side launch: pack weights (bf16) -> init state -> 50 x (stp+pack, gemm)
// ---------------------------------------------------------------------------
extern "C" void kernel_launch(void* const* d_in, const int* in_sizes, int n_in,
                              void* d_out, int out_size, void* d_ws, size_t ws_size,
                              hipStream_t stream) {
    (void)in_sizes; (void)n_in; (void)out_size; (void)ws_size;
    const float* ff       = (const float*)d_in[0];  // [B,T,N]
    const float* Wab      = (const float*)d_in[1];  // [N,N]
    const float* Wstp     = (const float*)d_in[2];  // [NA,NA]
    const float* rec_init = (const float*)d_in[3];  // [2,B,N]
    const float* Jstp     = (const float*)d_in[4];  // [1]
    float* out = (float*)d_out;                     // [B,T,N]

    char* ws = (char*)d_ws;
    size_t off = 0;
    unsigned short* Wab_p  = (unsigned short*)(ws + off); off += (size_t)NN * NN * 2;
    unsigned short* Wstp_p = (unsigned short*)(ws + off); off += (size_t)NA * NA * 2;
    float* rates = (float*)(ws + off); off += (size_t)BB * NN * 4;
    float* rec0  = (float*)(ws + off); off += (size_t)BB * NN * 4;
    float* rec1  = (float*)(ws + off); off += (size_t)BB * NN * 4;
    float* u     = (float*)(ws + off); off += (size_t)BB * NA * 4;
    float* x     = (float*)(ws + off); off += (size_t)BB * NA * 4;
    unsigned short* Ar = (unsigned short*)(ws + off); off += (size_t)BB * NN * 2;
    unsigned short* Aa = (unsigned short*)(ws + off); off += (size_t)BB * NA * 2;

    pack_weights<<<((size_t)NN * NN + 255) / 256, 256, 0, stream>>>(Wab, Wab_p, NN, NN, KT_AB);
    pack_weights<<<((size_t)NA * NA + 255) / 256, 256, 0, stream>>>(Wstp, Wstp_p, NA, NA, KT_ST);

    init_state<<<(BB * NN + 255) / 256, 256, 0, stream>>>(ff, rec_init, rates, rec0, rec1, u, x);

    for (int t = 0; t < TT; ++t) {
        stp_and_pack<<<(BB * NN + 255) / 256, 256, 0, stream>>>(rates, u, x, Ar, Aa);
        gemm_step<<<(NN / 16) / 4, 128, 262144, stream>>>(Wab_p, Wstp_p, Ar, Aa, ff, Jstp,
                                                          rates, rec0, rec1, out, t);
    }
}
